// ContinuousTimeAttention_16303695855792
// MI455X (gfx1250) — compile-verified, matrix operand order checked by reading, not run on hardware
//
#include <hip/hip_runtime.h>
#include <hip/hip_bf16.h>

// ---------------------------------------------------------------------------
// ContinuousTimeAttention for MI455X (gfx1250, wave32, WMMA bf16)
// ---------------------------------------------------------------------------
// Pipeline:
//   1) cvt f32->bf16 for x, Wq, Wk, Wv, Wo
//   2) WMMA GEMM with double-buffered async-to-LDS staging
//      (global_load_async_to_lds_b128 + s_wait_asynccnt):
//      Q,K bf16 [B,S,HID]; V stored transposed as Vt [B,NH,HD,S]
//   3) flash-attention, 32 queries/wave, multiplicative exp(-lambda*|dt|)
//      decay, causal mask, online softmax; WMMA for QK^T and P*V
//   4) WMMA GEMM: out = O@Wo^T + bo (f32)
// ---------------------------------------------------------------------------

typedef __attribute__((ext_vector_type(16))) __bf16 v16bf;
typedef __attribute__((ext_vector_type(8)))  float  v8f;

#define WMMA_BF16(a, b, c) \
  __builtin_amdgcn_wmma_f32_16x16x32_bf16(false, (a), false, (b), (short)0, (c), false, false)

union Frag {
  v16bf v;
  uint4 u[2];
};

__device__ __forceinline__ __bf16 tobf(float f) {
  union { float f; unsigned u; } a;
  a.f = f;
  unsigned u = a.u + 0x7FFFu + ((a.u >> 16) & 1u);   // round-to-nearest-even
  union { unsigned short s; __bf16 b; } r;
  r.s = (unsigned short)(u >> 16);
  return r.b;
}

// Async DMA of 16 bytes global -> LDS, tracked by ASYNCcnt (CDNA5).
__device__ __forceinline__ void async_copy16(uint32_t lds_off,
                                             const void* gptr) {
  asm volatile("global_load_async_to_lds_b128 %0, %1, off"
               :
               : "v"(lds_off), "v"((uint64_t)(uintptr_t)gptr)
               : "memory");
}

constexpr int Bc   = 2;
constexpr int Sc   = 2048;
constexpr int HIDc = 1024;
constexpr int NHc  = 16;
constexpr int HDc  = 64;

// ---------------------------------------------------------------------------
// f32 -> bf16 elementwise conversion
// ---------------------------------------------------------------------------
__global__ void cvt_bf16_kernel(const float* __restrict__ in,
                                __bf16* __restrict__ out, int n) {
  int i = blockIdx.x * blockDim.x + threadIdx.x;
  if (i < n) out[i] = tobf(in[i]);
}

// ---------------------------------------------------------------------------
// GEMM: C[m,n] = sum_k A[m,k] * W[n,k] + bias[n]
//   A: [M,K] bf16 row-major, W: [N,K] bf16 row-major (Linear weight layout)
//   MODE 0: store f32 row-major [M,N]
//   MODE 1: store bf16 row-major [M,N]
//   MODE 2: store bf16 transposed per-head: out[((b*NH+h)*HD+d)*S + s]
// Block = 128 threads (4 waves, 2x2 wave grid) computing a 64x128 tile.
// Per 32-wide k-step the block stages A(64x32) and B(128x32) into LDS with
// async-to-LDS DMA (double buffered), then each wave runs 8 WMMAs on its
// 32x64 register tile from LDS fragments.
// ---------------------------------------------------------------------------
template <int MODE>
__global__ __launch_bounds__(128) void gemm16_kernel(
    const __bf16* __restrict__ A, const __bf16* __restrict__ W,
    const float* __restrict__ bias, void* __restrict__ outp,
    int M, int N, int K) {
  __shared__ alignas(16) __bf16 At[2][64][32];    // 2 x 4 KB
  __shared__ alignas(16) __bf16 Bt[2][128][32];   // 2 x 8 KB

  const int tid  = threadIdx.x;
  const int lane = tid & 31;
  const int wv   = tid >> 5;
  const int ln   = lane & 15;
  const int hi   = lane >> 4;
  const int mblk = blockIdx.x * 64;
  const int nblk = blockIdx.y * 128;
  const int m0w  = (wv >> 1) * 32;   // wave row offset inside block tile
  const int n0w  = (wv & 1) * 64;    // wave col offset inside block tile

  // Cooperative async staging: 16B chunks, 4 chunks per 32-elem row.
  auto issue = [&](int buf, int k0) {
#pragma unroll
    for (int rd = 0; rd < 2; rd++) {           // A: 64*32*2B / (128*16B) = 2
      const int c = rd * 128 + tid;
      const int row = c >> 2, ko = (c & 3) * 8;
      async_copy16((uint32_t)(uintptr_t)&At[buf][row][ko],
                   A + (size_t)(mblk + row) * K + k0 + ko);
    }
#pragma unroll
    for (int rd = 0; rd < 4; rd++) {           // B: 128*32*2B / (128*16B) = 4
      const int c = rd * 128 + tid;
      const int row = c >> 2, ko = (c & 3) * 8;
      async_copy16((uint32_t)(uintptr_t)&Bt[buf][row][ko],
                   W + (size_t)(nblk + row) * K + k0 + ko);
    }
  };

  v8f c[2][4];
#pragma unroll
  for (int i = 0; i < 2; i++)
#pragma unroll
    for (int j = 0; j < 4; j++) c[i][j] = (v8f){};

  issue(0, 0);  // prologue: 6 async loads outstanding per wave
  int buf = 0;
  for (int k0 = 0; k0 < K; k0 += 32) {
    const bool more = (k0 + 32 < K);
    if (more) {
      issue(buf ^ 1, k0 + 32);
      // previous 6 transfers done; 6 for the next buffer stay in flight
      asm volatile("s_wait_asynccnt 0x6" ::: "memory");
    } else {
      asm volatile("s_wait_asynccnt 0x0" ::: "memory");
    }
    __syncthreads();

    Frag am[2], bn[4];
#pragma unroll
    for (int i = 0; i < 2; i++) {
      am[i].u[0] = *(const uint4*)(&At[buf][m0w + i * 16 + ln][hi * 8]);
      am[i].u[1] = *(const uint4*)(&At[buf][m0w + i * 16 + ln][16 + hi * 8]);
    }
#pragma unroll
    for (int j = 0; j < 4; j++) {
      bn[j].u[0] = *(const uint4*)(&Bt[buf][n0w + j * 16 + ln][hi * 8]);
      bn[j].u[1] = *(const uint4*)(&Bt[buf][n0w + j * 16 + ln][16 + hi * 8]);
    }
#pragma unroll
    for (int i = 0; i < 2; i++)
#pragma unroll
      for (int j = 0; j < 4; j++)
        c[i][j] = WMMA_BF16(am[i].v, bn[j].v, c[i][j]);

    __syncthreads();  // all waves done reading before this buffer is refilled
    buf ^= 1;
  }

  const int m0 = mblk + m0w;
  const int n0 = nblk + n0w;
#pragma unroll
  for (int i = 0; i < 2; i++) {
#pragma unroll
    for (int j = 0; j < 4; j++) {
      const int   col = n0 + j * 16 + ln;
      const float bb  = bias[col];
#pragma unroll
      for (int r = 0; r < 8; r++) {
        const int   row = m0 + i * 16 + r + hi * 8;
        const float val = c[i][j][r] + bb;
        if (MODE == 0) {
          ((float*)outp)[(size_t)row * N + col] = val;
        } else if (MODE == 1) {
          ((__bf16*)outp)[(size_t)row * N + col] = tobf(val);
        } else {
          const int bidx = row / Sc, s = row % Sc;
          const int h = col / HDc, d = col % HDc;
          ((__bf16*)outp)[(((size_t)bidx * NHc + h) * HDc + d) * (size_t)Sc +
                          s] = tobf(val);
        }
      }
    }
  }
}

// ---------------------------------------------------------------------------
// Flash attention with continuous-time multiplicative decay.
// One wave (block of 32) per (b, h, 32-query tile); keys processed 32/iter.
// Two 16-row query sub-tiles share every K/V fragment load.
// ---------------------------------------------------------------------------
__global__ __launch_bounds__(32) void attn_kernel(
    const __bf16* __restrict__ Q,   // [B,S,HID] bf16
    const __bf16* __restrict__ Km,  // [B,S,HID] bf16
    const __bf16* __restrict__ Vt,  // [B,NH,HD,S] bf16
    const float* __restrict__ tdl,  // time_deltas [B,S]
    const float* __restrict__ tdec, // time_decay [NH]
    __bf16* __restrict__ O) {       // [B,S,HID] bf16
  __shared__ alignas(16) __bf16 pl[2][16][32];  // P tiles for frag transpose

  const int lane = threadIdx.x;
  const int ln = lane & 15, hi = lane >> 4;
  const int qb = blockIdx.x * 32;
  const int h  = blockIdx.y;
  const int b  = blockIdx.z;

  const float td   = tdec[h];
  const float inv8 = 0.125f;  // 1/sqrt(HD)
  const float NEGI = -__builtin_inff();

  // Q fragments: [qtile][d-chunk] (A layout: lane = query row)
  Frag aq[2][2];
  float tq[2][8];
  int   qrow[2][8];
#pragma unroll
  for (int qt = 0; qt < 2; qt++) {
    const __bf16* qp =
        Q + ((size_t)b * Sc + qb + qt * 16 + ln) * HIDc + h * HDc;
    aq[qt][0].u[0] = *(const uint4*)(qp + hi * 8);
    aq[qt][0].u[1] = *(const uint4*)(qp + 16 + hi * 8);
    aq[qt][1].u[0] = *(const uint4*)(qp + 32 + hi * 8);
    aq[qt][1].u[1] = *(const uint4*)(qp + 48 + hi * 8);
#pragma unroll
    for (int r = 0; r < 8; r++) {
      qrow[qt][r] = qb + qt * 16 + r + hi * 8;
      tq[qt][r]   = tdl[(size_t)b * Sc + qrow[qt][r]];
    }
  }

  v8f acc[2][4];
  float mrow[2][8], lrow[2][8];
#pragma unroll
  for (int qt = 0; qt < 2; qt++) {
#pragma unroll
    for (int t = 0; t < 4; t++) acc[qt][t] = (v8f){};
#pragma unroll
    for (int r = 0; r < 8; r++) { mrow[qt][r] = NEGI; lrow[qt][r] = 0.f; }
  }

  const __bf16* vbase = Vt + ((size_t)(b * NHc + h)) * HDc * Sc;
  const int nkb = (qb + 31) / 32 + 1;  // key blocks under the causal mask

  for (int kb = 0; kb < nkb; kb++) {
    const int j0 = kb * 32;

    // K fragments: [jtile][d-chunk] (B layout: lane = key column j)
    Frag bk[2][2];
#pragma unroll
    for (int jt = 0; jt < 2; jt++) {
      const __bf16* kp =
          Km + ((size_t)b * Sc + j0 + jt * 16 + ln) * HIDc + h * HDc;
      bk[jt][0].u[0] = *(const uint4*)(kp + hi * 8);
      bk[jt][0].u[1] = *(const uint4*)(kp + 16 + hi * 8);
      bk[jt][1].u[0] = *(const uint4*)(kp + 32 + hi * 8);
      bk[jt][1].u[1] = *(const uint4*)(kp + 48 + hi * 8);
    }
    if (kb + 1 < nkb) {  // prefetch next key block (global_prefetch_b8)
      __builtin_prefetch(
          Km + ((size_t)b * Sc + j0 + 32 + ln) * HIDc + h * HDc, 0, 1);
      __builtin_prefetch(vbase + (size_t)ln * Sc + j0 + 32, 0, 1);
    }

    // scores: 8 WMMAs for the 2x2 grid of 16x16 tiles
    v8f sc[2][2];
#pragma unroll
    for (int qt = 0; qt < 2; qt++)
#pragma unroll
      for (int jt = 0; jt < 2; jt++) {
        v8f z = {};
        z = WMMA_BF16(aq[qt][0].v, bk[jt][0].v, z);
        z = WMMA_BF16(aq[qt][1].v, bk[jt][1].v, z);
        sc[qt][jt] = z;
      }

    const float tk0 = tdl[(size_t)b * Sc + j0 + ln];
    const float tk1 = tdl[(size_t)b * Sc + j0 + 16 + ln];

#pragma unroll
    for (int qt = 0; qt < 2; qt++) {
      float s0[8], s1[8];
#pragma unroll
      for (int r = 0; r < 8; r++) {
        const float d0 = __expf(-td * fabsf(tq[qt][r] - tk0));
        const float d1 = __expf(-td * fabsf(tq[qt][r] - tk1));
        // multiplicative decay on scores, then causal mask (keep j <= q)
        s0[r] = (j0 + ln      <= qrow[qt][r]) ? sc[qt][0][r] * inv8 * d0 : NEGI;
        s1[r] = (j0 + 16 + ln <= qrow[qt][r]) ? sc[qt][1][r] * inv8 * d1 : NEGI;
      }
#pragma unroll
      for (int r = 0; r < 8; r++) {
        // row max across the 16 lanes of this half
        float v = fmaxf(s0[r], s1[r]);
        v = fmaxf(v, __shfl_xor(v, 1, 32));
        v = fmaxf(v, __shfl_xor(v, 2, 32));
        v = fmaxf(v, __shfl_xor(v, 4, 32));
        v = fmaxf(v, __shfl_xor(v, 8, 32));
        const float mn    = fmaxf(mrow[qt][r], v);
        const float scale = __expf(mrow[qt][r] - mn);
        mrow[qt][r] = mn;
        const float p0 = __expf(s0[r] - mn);
        const float p1 = __expf(s1[r] - mn);
        float rs = p0 + p1;
        rs += __shfl_xor(rs, 1, 32);
        rs += __shfl_xor(rs, 2, 32);
        rs += __shfl_xor(rs, 4, 32);
        rs += __shfl_xor(rs, 8, 32);
        lrow[qt][r] = lrow[qt][r] * scale + rs;
        acc[qt][0][r] *= scale;
        acc[qt][1][r] *= scale;
        acc[qt][2][r] *= scale;
        acc[qt][3][r] *= scale;
        // C-layout (lane=col) -> LDS so we can reload in A-layout (lane=row)
        pl[qt][r + hi * 8][ln]      = tobf(p0);
        pl[qt][r + hi * 8][16 + ln] = tobf(p1);
      }
    }
    __syncthreads();

    Frag apf[2];
#pragma unroll
    for (int qt = 0; qt < 2; qt++) {
      apf[qt].u[0] = *(const uint4*)(&pl[qt][ln][hi * 8]);
      apf[qt].u[1] = *(const uint4*)(&pl[qt][ln][16 + hi * 8]);
    }
    __syncthreads();

    // P(16x32) x V(32x64): Vt rows are contiguous in j
    Frag bv[4];
#pragma unroll
    for (int t = 0; t < 4; t++) {
      const __bf16* vp = vbase + (size_t)(t * 16 + ln) * Sc + j0;
      bv[t].u[0] = *(const uint4*)(vp + hi * 8);
      bv[t].u[1] = *(const uint4*)(vp + 16 + hi * 8);
    }
#pragma unroll
    for (int qt = 0; qt < 2; qt++)
#pragma unroll
      for (int t = 0; t < 4; t++)
        acc[qt][t] = WMMA_BF16(apf[qt].v, bv[t].v, acc[qt][t]);
  }

  // normalize and store O (bf16 [B,S,HID]; head-major cols -> row-major HID)
#pragma unroll
  for (int qt = 0; qt < 2; qt++) {
#pragma unroll
    for (int r = 0; r < 8; r++) {
      const float invl = 1.f / lrow[qt][r];
      const size_t ob = ((size_t)b * Sc + qrow[qt][r]) * HIDc + h * HDc + ln;
      O[ob]      = tobf(acc[qt][0][r] * invl);
      O[ob + 16] = tobf(acc[qt][1][r] * invl);
      O[ob + 32] = tobf(acc[qt][2][r] * invl);
      O[ob + 48] = tobf(acc[qt][3][r] * invl);
    }
  }
}

// ---------------------------------------------------------------------------
extern "C" void kernel_launch(void* const* d_in, const int* in_sizes, int n_in,
                              void* d_out, int out_size, void* d_ws,
                              size_t ws_size, hipStream_t stream) {
  const float* x    = (const float*)d_in[0];
  const float* tdl  = (const float*)d_in[1];
  const float* Wq   = (const float*)d_in[2];
  const float* bq   = (const float*)d_in[3];
  const float* Wk   = (const float*)d_in[4];
  const float* bk   = (const float*)d_in[5];
  const float* Wv   = (const float*)d_in[6];
  const float* bv   = (const float*)d_in[7];
  const float* Wo   = (const float*)d_in[8];
  const float* bo   = (const float*)d_in[9];
  const float* tdec = (const float*)d_in[10];
  float* out = (float*)d_out;

  const int nX = Bc * Sc * HIDc;  // 4M elements
  const int nW = HIDc * HIDc;     // 1M elements

  char* ws = (char*)d_ws;
  __bf16* xb  = (__bf16*)ws; ws += (size_t)nX * 2;
  __bf16* wqb = (__bf16*)ws; ws += (size_t)nW * 2;
  __bf16* wkb = (__bf16*)ws; ws += (size_t)nW * 2;
  __bf16* wvb = (__bf16*)ws; ws += (size_t)nW * 2;
  __bf16* wob = (__bf16*)ws; ws += (size_t)nW * 2;
  __bf16* Qb  = (__bf16*)ws; ws += (size_t)nX * 2;
  __bf16* Kb  = (__bf16*)ws; ws += (size_t)nX * 2;
  __bf16* Vtb = (__bf16*)ws; ws += (size_t)nX * 2;
  __bf16* Ob  = (__bf16*)ws; ws += (size_t)nX * 2;  // total 48 MB

  // 1) conversions
  cvt_bf16_kernel<<<(nX + 255) / 256, 256, 0, stream>>>(x, xb, nX);
  cvt_bf16_kernel<<<(nW + 255) / 256, 256, 0, stream>>>(Wq, wqb, nW);
  cvt_bf16_kernel<<<(nW + 255) / 256, 256, 0, stream>>>(Wk, wkb, nW);
  cvt_bf16_kernel<<<(nW + 255) / 256, 256, 0, stream>>>(Wv, wvb, nW);
  cvt_bf16_kernel<<<(nW + 255) / 256, 256, 0, stream>>>(Wo, wob, nW);

  // 2) projections (M=4096, N=1024, K=1024); block tiles 64x128
  const int M = Bc * Sc;
  dim3 gg(M / 64, HIDc / 128);
  gemm16_kernel<1><<<gg, 128, 0, stream>>>(xb, wqb, bq, Qb, M, HIDc, HIDc);
  gemm16_kernel<1><<<gg, 128, 0, stream>>>(xb, wkb, bk, Kb, M, HIDc, HIDc);
  gemm16_kernel<2><<<gg, 128, 0, stream>>>(xb, wvb, bv, Vtb, M, HIDc, HIDc);

  // 3) attention: 32 queries per wave
  attn_kernel<<<dim3(Sc / 32, NHc, Bc), 32, 0, stream>>>(Qb, Kb, Vtb, tdl,
                                                         tdec, Ob);

  // 4) output projection -> f32
  gemm16_kernel<0><<<gg, 128, 0, stream>>>(Ob, wob, bo, out, M, HIDc, HIDc);
}
